// NeighborhoodAggr_86560770883758
// MI455X (gfx1250) — matrix operand order repaired and checked by reading, not compiled, not run on hardware
//
#include <hip/hip_runtime.h>
#include <hip/hip_bf16.h>
#include <math.h>

// ---- CDNA5 WMMA vector types ----
typedef _Float16 v16h __attribute__((ext_vector_type(16)));
typedef _Float16 v8h  __attribute__((ext_vector_type(8)));
typedef float    v8f  __attribute__((ext_vector_type(8)));

#define HIDDEN   128
#define HEADS    8
#define HEAD_DIM 16
#define TDIM     64
#define RDIM     64
#define NNB      64     // neighbors per node
#define START_T  0.2f
#define END_T    0.8f
#define NORM     0.25f  // 1/sqrt(16)

// Workspace layout:
//   [0, 65536)          : _Float16 wfrag[16 ntile][4 kstep][32 lane][16 e]  (permuted W, f16)
//   [65536, 66560)      : float biasKV[256]   (rb_k+tb_k | rb_v+tb_v)
//   [66560, 67072)      : float qconst[128]   (time2vec(START_T) @ tW_q + tb_q)
#define WS_WFRAG_BYTES  65536
#define WS_BIAS_OFF     65536
#define WS_QCONST_OFF   (65536 + 1024)

// ---------------------------------------------------------------------------
// Prep kernel: permute combined weight [128 x 256] into WMMA B-fragment order,
// fold biases, and compute the constant q time-offset vector.
// B packing used here (contiguous k, A matches): lane l holds column N = (l&15);
//   element e -> K = kstep*32 + (l>>4)*16 + e.
// ---------------------------------------------------------------------------
__global__ void naggr_prep_kernel(
    const float* __restrict__ t2v_w, const float* __restrict__ t2v_b,
    const float* __restrict__ tW_k,  const float* __restrict__ tb_k,
    const float* __restrict__ tW_q,  const float* __restrict__ tb_q,
    const float* __restrict__ tW_v,  const float* __restrict__ tb_v,
    const float* __restrict__ rW_k,  const float* __restrict__ rb_k,
    const float* __restrict__ rW_v,  const float* __restrict__ rb_v,
    _Float16* __restrict__ wfrag, float* __restrict__ biasKV,
    float* __restrict__ qconst)
{
    int idx = blockIdx.x * 256 + threadIdx.x;
    if (idx < 32768) {
        // idx = ((ntile*4 + kstep)*32 + lane)*16 + e
        int e     = idx & 15;
        int lane  = (idx >> 4) & 31;
        int f     = idx >> 9;
        int ntile = f >> 2;
        int ks    = f & 3;
        int K   = ks * 32 + (lane >> 4) * 16 + e;   // 0..127 (rels rows | t2v rows)
        int col = ntile * 16 + (lane & 15);         // 0..255 (K-proj | V-proj)
        float val;
        if (col < HIDDEN) {
            val = (K < RDIM) ? rW_k[K * HIDDEN + col]
                             : tW_k[(K - RDIM) * HIDDEN + col];
        } else {
            int c = col - HIDDEN;
            val = (K < RDIM) ? rW_v[K * HIDDEN + c]
                             : tW_v[(K - RDIM) * HIDDEN + c];
        }
        wfrag[idx] = (_Float16)val;
    } else if (idx < 32768 + 256) {
        int c = idx - 32768;
        biasKV[c] = (c < HIDDEN) ? (rb_k[c] + tb_k[c])
                                 : (rb_v[c - HIDDEN] + tb_v[c - HIDDEN]);
    } else if (idx < 32768 + 256 + HIDDEN) {
        int c = idx - 32768 - 256;
        float acc = tb_q[c];
        for (int j = 0; j < TDIM; ++j) {
            float fj = START_T * t2v_w[j] + t2v_b[j];
            float tv = (j == 0) ? fj : __sinf(fj);
            acc += tv * tW_q[j * HIDDEN + c];
        }
        qconst[c] = acc;
    }
}

// ---------------------------------------------------------------------------
// Fused per-target kernel: 1 block = 1 target node, 256 threads = 8 wave32.
// Phase 1: build X = [rels | time2vec] (64x128 f16) in LDS.
// Phase 2: WMMA GEMM X(64x128) @ W(128x256) -> [k_proj | v_proj].
// Phase 3: add gathered k_tab/v_tab rows + bias, store k/v (f16) in LDS.
// Phase 4: masked 8-head attention, softmax over 64 neighbors, output.
// ---------------------------------------------------------------------------
__global__ __launch_bounds__(256)
void naggr_main_kernel(
    const float* __restrict__ k_tab, const float* __restrict__ q_tab,
    const float* __restrict__ v_tab, const int* __restrict__ nid,
    const int* __restrict__ neigh,   const float* __restrict__ times,
    const float* __restrict__ rels,  const float* __restrict__ t2v_w,
    const float* __restrict__ t2v_b, const _Float16* __restrict__ wfrag,
    const float* __restrict__ biasKV,const float* __restrict__ qconst,
    float* __restrict__ out)
{
    __shared__ _Float16 s_kv[NNB * 256];   // phase1-2: X[64][128] / phase3+: kv[64][256]
    __shared__ float s_sc[HEADS * NNB];    // scores -> attention weights
    __shared__ float s_q[HIDDEN];
    __shared__ float s_t[NNB];
    __shared__ int   s_nb[NNB];
    __shared__ float s_red[17];            // [0..7]=max, [8..15]=sum, [16]=any_valid

    const int b   = blockIdx.x;
    const int tid = threadIdx.x;
    _Float16* X = s_kv;                    // alias; X dead after A-frag load

    if (tid < NNB) {
        s_t[tid]  = times[(size_t)b * NNB + tid];
        s_nb[tid] = neigh[(size_t)b * NNB + tid];
    }
    if (tid < HIDDEN)
        s_q[tid] = q_tab[(size_t)nid[b] * HIDDEN + tid] + qconst[tid];
    __syncthreads();

    // ---- Phase 1: X = [rels | t2v(times)], f16, row-major 64x128 ----
    for (int idx = tid; idx < NNB * HIDDEN; idx += 256) {
        int n = idx >> 7, c = idx & 127;
        float val;
        if (c < RDIM) {
            val = rels[((size_t)b * NNB + n) * RDIM + c];
        } else {
            int j = c - RDIM;
            float f = s_t[n] * t2v_w[j] + t2v_b[j];
            val = (j == 0) ? f : __sinf(f);
        }
        X[idx] = (_Float16)val;
    }
    __syncthreads();

    // ---- Phase 2: WMMA GEMM ----
    const int wave = tid >> 5, lane = tid & 31;
    const int mt    = wave & 3;            // M-tile 0..3 (rows of 16 neighbors)
    const int nbase = (wave >> 2) * 8;     // N-tiles: waves 0-3 -> 0..7, 4-7 -> 8..15
    const int l15 = lane & 15, hlf = lane >> 4;

    // A-fragment (16x32 f16, matches B packing): lane row = mt*16+l15;
    //   e<8 : K = ks*32 + hlf*16 + e       (contiguous 8 halves)
    //   e>=8: K = ks*32 + hlf*16 + e       (contiguous 8 halves)
    v16h afrag[4];
    {
        int row = mt * 16 + l15;
        #pragma unroll
        for (int ks = 0; ks < 4; ++ks) {
            int cbase = ks * 32 + hlf * 16;
            v8h lo = *(const v8h*)&X[row * HIDDEN + cbase];
            v8h hi = *(const v8h*)&X[row * HIDDEN + cbase + 8];
            afrag[ks] = __builtin_shufflevector(lo, hi,
                0,1,2,3,4,5,6,7,8,9,10,11,12,13,14,15);
        }
    }
    __syncthreads();   // X fully consumed; LDS now reusable as kv

    v8f acc[8] = {};
    #pragma unroll
    for (int t = 0; t < 8; ++t) {
        int ntile = nbase + t;
        #pragma unroll
        for (int ks = 0; ks < 4; ++ks) {
            v16h bfrag = *((const v16h*)wfrag + ((ntile * 4 + ks) * 32 + lane));
            acc[t] = __builtin_amdgcn_wmma_f32_16x16x32_f16(
                false, afrag[ks], false, bfrag, (short)0, acc[t], false, false);
        }
    }

    // ---- Phase 3: epilogue — add gathered table rows + bias, store f16 kv ----
    // C/D layout: row = mt*16 + hlf*8 + v, col = ntile*16 + l15.
    #pragma unroll
    for (int t = 0; t < 8; ++t) {
        int ntile = nbase + t;
        int col = ntile * 16 + l15;         // 0..255
        float bias = biasKV[col];
        #pragma unroll
        for (int v = 0; v < 8; ++v) {
            int row = mt * 16 + hlf * 8 + v;
            size_t nb = (size_t)s_nb[row];
            float tabv = (col < HIDDEN) ? k_tab[nb * HIDDEN + col]
                                        : v_tab[nb * HIDDEN + (col - HIDDEN)];
            s_kv[row * 256 + col] = (_Float16)(acc[t][v] + tabv + bias);
        }
    }
    __syncthreads();

    // ---- Phase 4a: masked scores ----
    for (int p = tid; p < HEADS * NNB; p += 256) {
        int h = p >> 6, n = p & 63;
        float d = 0.0f;
        #pragma unroll
        for (int dd = 0; dd < HEAD_DIM; ++dd)
            d += s_q[h * HEAD_DIM + dd] * (float)s_kv[n * 256 + h * HEAD_DIM + dd];
        float tn = s_t[n];
        bool valid = (tn >= START_T) && (tn < END_T);
        s_sc[h * NNB + n] = valid ? d * NORM : -1.0e30f;
    }
    __syncthreads();

    // ---- Phase 4b: per-head max / sum-of-exp, plus any_valid ----
    if (tid < HEADS) {
        int h = tid;
        float m = -3.4e38f;
        for (int n = 0; n < NNB; ++n) m = fmaxf(m, s_sc[h * NNB + n]);
        float ssum = 0.0f;
        for (int n = 0; n < NNB; ++n) ssum += __expf(s_sc[h * NNB + n] - m);
        s_red[h] = m;
        s_red[8 + h] = ssum;
    }
    if (tid == 8) {
        float anyv = 0.0f;
        for (int n = 0; n < NNB; ++n) {
            float tn = s_t[n];
            if (tn >= START_T && tn < END_T) anyv = 1.0f;
        }
        s_red[16] = anyv;
    }
    __syncthreads();

    // ---- Phase 4c: normalize attention in place ----
    for (int p = tid; p < HEADS * NNB; p += 256) {
        int h = p >> 6;
        s_sc[p] = __expf(s_sc[p] - s_red[h]) / s_red[8 + h];
    }
    __syncthreads();

    // ---- Phase 4d: weighted value sum -> output ----
    if (tid < HIDDEN) {
        int c = tid, h = c >> 4;
        float o = 0.0f;
        for (int n = 0; n < NNB; ++n)
            o += s_sc[h * NNB + n] * (float)s_kv[n * 256 + HIDDEN + c];
        out[(size_t)b * HIDDEN + c] = (s_red[16] != 0.0f) ? o : 0.0f;
    }
}

extern "C" void kernel_launch(void* const* d_in, const int* in_sizes, int n_in,
                              void* d_out, int out_size, void* d_ws, size_t ws_size,
                              hipStream_t stream) {
    const float* k_tab = (const float*)d_in[0];
    const float* q_tab = (const float*)d_in[1];
    const float* v_tab = (const float*)d_in[2];
    const int*   nid   = (const int*)  d_in[3];
    const int*   neigh = (const int*)  d_in[4];
    const float* times = (const float*)d_in[5];
    const float* rels  = (const float*)d_in[6];
    const float* t2v_w = (const float*)d_in[7];
    const float* t2v_b = (const float*)d_in[8];
    const float* tW_k  = (const float*)d_in[9];
    const float* tb_k  = (const float*)d_in[10];
    const float* tW_q  = (const float*)d_in[11];
    const float* tb_q  = (const float*)d_in[12];
    const float* tW_v  = (const float*)d_in[13];
    const float* tb_v  = (const float*)d_in[14];
    const float* rW_k  = (const float*)d_in[15];
    const float* rb_k  = (const float*)d_in[16];
    const float* rW_v  = (const float*)d_in[17];
    const float* rb_v  = (const float*)d_in[18];

    _Float16* wfrag = (_Float16*)d_ws;
    float* biasKV   = (float*)((char*)d_ws + WS_BIAS_OFF);
    float* qconst   = (float*)((char*)d_ws + WS_QCONST_OFF);

    const int Bn = in_sizes[3];   // number of target nodes (4096)

    // Prep: 32768 wfrag elems + 256 biases + 128 qconst
    int prep_threads = 32768 + 256 + 128;
    naggr_prep_kernel<<<(prep_threads + 255) / 256, 256, 0, stream>>>(
        t2v_w, t2v_b, tW_k, tb_k, tW_q, tb_q, tW_v, tb_v,
        rW_k, rb_k, rW_v, rb_v, wfrag, biasKV, qconst);

    naggr_main_kernel<<<Bn, 256, 0, stream>>>(
        k_tab, q_tab, v_tab, nid, neigh, times, rels, t2v_w, t2v_b,
        wfrag, biasKV, qconst, (float*)d_out);
}
